// Decoder_53532472377927
// MI455X (gfx1250) — matrix operand order repaired and checked by reading, not compiled, not run on hardware
//
#include <hip/hip_runtime.h>
#include <hip/hip_bf16.h>
#include <math.h>

// ---------------- problem constants ----------------
#define B_        32
#define T_ENC_    300
#define T_DEC_    500
#define N_MEL_    80
#define ENC_DIM_  512
#define RNN_      1024
#define PRE_      256
#define ATT_DIM_  128
#define LOC_F_    32
#define LOC_K_    31
#define PAD_      15
#define NG_       4096            // 4*RNN gates
#define K_ATT_    1792            // PRE + ENC_DIM + RNN
#define K_DEC_    2560            // RNN + ENC_DIM + RNN
#define NKT_ATT_  (K_ATT_ / 32)   // 56
#define NKT_DEC_  (K_DEC_ / 32)   // 80

// output layout (floats)
#define GATE_OFF_  (B_ * N_MEL_ * T_DEC_)            // 1,280,000
#define ALIGN_OFF_ (GATE_OFF_ + B_ * T_DEC_)         // 1,296,000

// persistent grid
#define NWG_   32
#define NTHR_  (NWG_ * 256)                          // 8192 threads
#define NWAVE_ (NTHR_ / 32)                          // 256 waves

// workspace element counts
#define AATT_N     (B_ * K_ATT_)      // 57344 bf16 (packed A-frag layout)
#define ADEC_N     (B_ * K_DEC_)      // 81920 bf16
#define AC_N       (B_ * RNN_)
#define DC_N       (B_ * RNN_)
#define AW_N       (B_ * T_ENC_)
#define CTXF_N     (B_ * ENC_DIM_)
#define DHF_N      (B_ * RNN_)
#define QP_N       (B_ * ATT_DIM_)
#define GATES_N    (B_ * NG_)
// state region: bf16 packed A buffers first (32B aligned), then floats
#define STATE_F_N  (AC_N + DC_N + 3 * AW_N + QP_N + CTXF_N + DHF_N + GATES_N)
#define STATE_WORDS ((AATT_N + ADEC_N) / 2 + STATE_F_N)   // u32 words to zero

#define XPRE_N     (T_DEC_ * B_ * PRE_)              // 4,096,000 f32
#define PM_N       (B_ * T_ENC_ * ATT_DIM_)          // 1,228,800 f32
#define WATT_N     (NG_ * K_ATT_)                    // bf16
#define WDEC_N     (NG_ * K_DEC_)                    // bf16
#define WQ_N       (ATT_DIM_ * RNN_)                 // bf16

typedef __attribute__((ext_vector_type(16))) __bf16 v16bf;
typedef __attribute__((ext_vector_type(8)))  float  v8f;

__device__ __forceinline__ float sigf(float x) { return 1.0f / (1.0f + expf(-x)); }

__device__ __forceinline__ float drop_mask(unsigned seed) {
    unsigned h = seed * 0x9E3779B9u;
    h ^= h >> 16; h *= 0x7FEB352Du;
    h ^= h >> 15; h *= 0x846CA68Bu;
    h ^= h >> 16;
    float u = (float)(h >> 8) * (1.0f / 16777216.0f);
    return (u < 0.1f) ? 0.0f : (1.0f / 0.9f);
}

// Index into packed WMMA A-fragment buffer [mt][kt][lane][16] for element (row b, col k).
// 16-bit A 16x32 layout: lane = (b&15) + 16*((k>>3)&1), elem j = (k&7) + 8*((k&31)>>4).
__device__ __forceinline__ int a_index(int b, int k, int nktA) {
    int mt = b >> 4, r = b & 15;
    int kt = k >> 5, rem = k & 31;
    int lane = r + (((rem >> 3) & 1) << 4);
    int j = (rem & 7) + (((rem >> 4) & 1) << 3);
    return ((mt * nktA + kt) * 32 + lane) * 16 + j;
}

// ---------------- grid-wide barrier (persistent kernel) ----------------
__device__ __forceinline__ void gridbar(unsigned* cnt, unsigned* gen) {
    __syncthreads();
    if (threadIdx.x == 0) {
        __threadfence();
        unsigned g = *(volatile unsigned*)gen;
        if (atomicAdd(cnt, 1u) == NWG_ - 1u) {
            *(volatile unsigned*)cnt = 0u;
            __threadfence();
            *(volatile unsigned*)gen = g + 1u;
        } else {
            while (*(volatile unsigned*)gen == g) { __builtin_amdgcn_s_sleep(1); }
        }
        __threadfence();
    }
    __syncthreads();
}

// ---------------- init: zero barrier + recurrent state (u32 words) ----------------
__global__ void init_kernel(unsigned* bar, unsigned* state, int nwords) {
    int i = blockIdx.x * blockDim.x + threadIdx.x;
    if (i < 2) bar[i] = 0u;
    for (int k = i; k < nwords; k += gridDim.x * blockDim.x) state[k] = 0u;
}

// ---------------- prenet over all timesteps (wave per (t,b) row) ----------------
__global__ __launch_bounds__(256) void prenet_kernel(const float* __restrict__ dec_in,
                                                     const float* __restrict__ W1,
                                                     const float* __restrict__ W2,
                                                     float* __restrict__ xpre) {
    __shared__ float sh[8 * (N_MEL_ + PRE_)];
    int w    = threadIdx.x >> 5;
    int lane = threadIdx.x & 31;
    int gw   = blockIdx.x * 8 + w;          // 0 .. 15999 exactly
    int t = gw >> 5, b = gw & 31;
    float* sx = sh + w * (N_MEL_ + PRE_);
    float* sH = sx + N_MEL_;
    for (int m = lane; m < N_MEL_; m += 32)
        sx[m] = (t == 0) ? 0.0f : dec_in[(b * N_MEL_ + m) * T_DEC_ + (t - 1)];
    __syncthreads();
    float h1[8];
#pragma unroll
    for (int i = 0; i < 8; ++i) {
        int u = lane + 32 * i;
        float acc = 0.0f;
        for (int m = 0; m < N_MEL_; ++m) acc += sx[m] * W1[u * N_MEL_ + m];
        acc = fmaxf(acc, 0.0f);
        acc *= drop_mask(((0u * T_DEC_ + t) * B_ + b) * PRE_ + u);
        h1[i] = acc;
    }
    __syncthreads();
#pragma unroll
    for (int i = 0; i < 8; ++i) sH[lane + 32 * i] = h1[i];
    __syncthreads();
#pragma unroll
    for (int i = 0; i < 8; ++i) {
        int u = lane + 32 * i;
        float acc = 0.0f;
        for (int m = 0; m < PRE_; ++m) acc += sH[m] * W2[u * PRE_ + m];
        acc = fmaxf(acc, 0.0f);
        acc *= drop_mask(0x80000000u + ((1u * T_DEC_ + t) * B_ + b) * PRE_ + u);
        xpre[((size_t)t * B_ + b) * PRE_ + u] = acc;
    }
}

// ---------------- processed_memory = memory @ m_W^T ----------------
__global__ void pm_kernel(const float* __restrict__ memory, const float* __restrict__ mW,
                          float* __restrict__ pm) {
    int idx = blockIdx.x * blockDim.x + threadIdx.x;      // 32*300*128 exact
    int d = idx & 127, bs = idx >> 7;
    const float* mr = memory + (size_t)bs * ENC_DIM_;
    const float* wr = mW + d * ENC_DIM_;
    float acc = 0.0f;
    for (int k = 0; k < ENC_DIM_; ++k) acc += mr[k] * wr[k];
    pm[idx] = acc;
}

// ---------------- pack [Wih | Whh] into bf16 WMMA B-fragment layout ----------------
// dst element e = ((nt*nkt + kt)*32 + lane)*16 + j holds W[n, k] with
// n = nt*16 + (lane&15), k = kt*32 + (lane>>4)*8 + (j&7) + (j>>3)*16   (B 32x16 16-bit layout)
__global__ void pack_weights(const float* __restrict__ Wih, const float* __restrict__ Whh,
                             int Kih, int K, int N, __bf16* __restrict__ dst) {
    int total = N * K;
    int nkt = K >> 5;
    for (int e = blockIdx.x * blockDim.x + threadIdx.x; e < total;
         e += gridDim.x * blockDim.x) {
        int tile = e >> 9, w = e & 511;
        int lane = w >> 4, j = w & 15;
        int nt = tile / nkt, kt = tile - nt * nkt;
        int n = nt * 16 + (lane & 15);
        int k = kt * 32 + (lane >> 4) * 8 + (j & 7) + ((j >> 3) << 4);
        float v = (k < Kih) ? Wih[(size_t)n * Kih + k]
                            : Whh[(size_t)n * (K - Kih) + (k - Kih)];
        dst[e] = (__bf16)v;
    }
}

// -------- WMMA skinny GEMM on pre-packed bf16 fragments: out[32,N] = A @ W^T (+bias) ----
template <bool BIAS>
__device__ __forceinline__ void wmma_gemm_p(const __bf16* __restrict__ Ap, int nktA, int nkt,
                                            const __bf16* __restrict__ Wp, int ntiles,
                                            float* __restrict__ out, int ldo,
                                            const float* __restrict__ b0,
                                            const float* __restrict__ b1,
                                            int gwave, int lane) {
    const int r0 = lane & 15;
    const int hi = lane >> 4;
    const __bf16* a0p = Ap + lane * 16;                          // mt = 0 (rows 0..15)
    const __bf16* a1p = Ap + (size_t)nktA * 512 + lane * 16;     // mt = 1 (rows 16..31)
    for (int nt = gwave; nt < ntiles; nt += NWAVE_) {
        v8f acc0 = {0.f, 0.f, 0.f, 0.f, 0.f, 0.f, 0.f, 0.f};
        v8f acc1 = {0.f, 0.f, 0.f, 0.f, 0.f, 0.f, 0.f, 0.f};
        const __bf16* wb = Wp + ((size_t)nt * nkt) * 512 + lane * 16;
        for (int kt = 0; kt < nkt; ++kt) {
            v16bf bf  = *(const v16bf*)(wb + (size_t)kt * 512);
            v16bf af0 = *(const v16bf*)(a0p + (size_t)kt * 512);
            v16bf af1 = *(const v16bf*)(a1p + (size_t)kt * 512);
            acc0 = __builtin_amdgcn_wmma_f32_16x16x32_bf16(false, af0, false, bf,
                                                           (short)0, acc0, false, false);
            acc1 = __builtin_amdgcn_wmma_f32_16x16x32_bf16(false, af1, false, bf,
                                                           (short)0, acc1, false, false);
        }
        int n = nt * 16 + r0;
        float bias = 0.0f;
        if (BIAS) bias = b0[n] + b1[n];
        int mb = hi * 8;
#pragma unroll
        for (int r = 0; r < 8; ++r) {
            out[(size_t)(mb + r) * ldo + n]      = acc0[r] + bias;
            out[(size_t)(mb + r + 16) * ldo + n] = acc1[r] + bias;
        }
    }
}

// ---------------- persistent sequential decoder ----------------
__global__ __launch_bounds__(256) void decoder_kernel(
    const float* __restrict__ memory, const int* __restrict__ lens,
    const float* __restrict__ att_bih, const float* __restrict__ att_bhh,
    const float* __restrict__ vw, const float* __restrict__ Wc,
    const float* __restrict__ Wld,
    const float* __restrict__ dec_bih, const float* __restrict__ dec_bhh,
    const float* __restrict__ projW, const float* __restrict__ projB,
    const float* __restrict__ gateW, const float* __restrict__ gateB,
    const float* __restrict__ xpre, const float* __restrict__ pm,
    const __bf16* __restrict__ Watt, const __bf16* __restrict__ Wdec,
    const __bf16* __restrict__ Wq,
    __bf16* __restrict__ stb, unsigned* __restrict__ bar, float* __restrict__ out) {

    __bf16* Aatt = stb;                               // packed A frags [2][56][32][16]
    __bf16* Adec = Aatt + AATT_N;                     // packed A frags [2][80][32][16]
    float* ac    = (float*)(Adec + ADEC_N);
    float* dc    = ac + AC_N;
    float* aw    = dc + DC_N;
    float* awc   = aw + AW_N;
    float* ebuf  = awc + AW_N;
    float* qproj = ebuf + AW_N;
    float* ctxf  = qproj + QP_N;                      // fp32 ctx for projection phase
    float* dhf   = ctxf + CTXF_N;                     // fp32 dh  for projection phase
    float* gates = dhf + DHF_N;
    unsigned* cnt = bar;
    unsigned* gen = bar + 1;

    const int tid  = blockIdx.x * blockDim.x + threadIdx.x;   // 0..8191
    const int lane = threadIdx.x & 31;
    const int gw   = blockIdx.x * 8 + (threadIdx.x >> 5);     // 0..255

    for (int t = 0; t < T_DEC_; ++t) {
        // P0: stage x_t (bf16, A-fragment layout) into Aatt[:, 0:256]
        {
            int b = tid >> 8, c = tid & 255;
            Aatt[a_index(b, c, NKT_ATT_)] =
                (__bf16)xpre[((size_t)t * B_ + b) * PRE_ + c];
        }
        gridbar(cnt, gen);

        // P1: attention LSTM gates = [x|ctx|ah] @ [Wih|Whh]^T + biases
        wmma_gemm_p<true>(Aatt, NKT_ATT_, NKT_ATT_, Watt, NG_ / 16, gates, NG_,
                          att_bih, att_bhh, gw, lane);
        gridbar(cnt, gen);

        // P2: attention LSTM pointwise -> ah (packed into both A buffers), ac
        for (int idx = tid; idx < B_ * RNN_; idx += NTHR_) {
            int b = idx >> 10, n = idx & 1023;
            const float* g = gates + (size_t)b * NG_;
            float i = sigf(g[n]), f = sigf(g[RNN_ + n]);
            float gg = tanhf(g[2 * RNN_ + n]), o = sigf(g[3 * RNN_ + n]);
            float c = f * ac[idx] + i * gg;
            ac[idx] = c;
            __bf16 h = (__bf16)(o * tanhf(c));
            Aatt[a_index(b, PRE_ + ENC_DIM_ + n, NKT_ATT_)] = h;
            Adec[a_index(b, n, NKT_DEC_)] = h;
        }
        gridbar(cnt, gen);

        // P3a: qproj = ah @ q_W^T (ah = first 32 k-tiles of Adec)
        wmma_gemm_p<false>(Adec, NKT_DEC_, RNN_ / 32, Wq, ATT_DIM_ / 16, qproj, ATT_DIM_,
                           nullptr, nullptr, gw, lane);
        gridbar(cnt, gen);

        // P3b: location conv + dense + energies (masked)
        for (int pos = tid; pos < B_ * T_ENC_; pos += NTHR_) {
            int b = pos / T_ENC_, s = pos - b * T_ENC_;
            float conv[LOC_F_];
#pragma unroll
            for (int f = 0; f < LOC_F_; ++f) conv[f] = 0.0f;
            for (int k = 0; k < LOC_K_; ++k) {
                int ss = s + k - PAD_;
                float a0 = 0.0f, a1 = 0.0f;
                if (ss >= 0 && ss < T_ENC_) {
                    a0 = aw[b * T_ENC_ + ss];
                    a1 = awc[b * T_ENC_ + ss];
                }
#pragma unroll
                for (int f = 0; f < LOC_F_; ++f)
                    conv[f] += a0 * Wc[f * 62 + k] + a1 * Wc[f * 62 + LOC_K_ + k];
            }
            float e = 0.0f;
            const float* pmr = pm + (size_t)pos * ATT_DIM_;
            const float* qp  = qproj + b * ATT_DIM_;
            for (int d = 0; d < ATT_DIM_; ++d) {
                float ld = 0.0f;
#pragma unroll
                for (int f = 0; f < LOC_F_; ++f) ld += conv[f] * Wld[d * LOC_F_ + f];
                e += vw[d] * tanhf(qp[d] + ld + pmr[d]);
            }
            ebuf[pos] = (s < lens[b]) ? e : -3.0e38f;
        }
        gridbar(cnt, gen);

        // P3c: masked softmax over T_ENC, one wave per batch row
        if (gw < B_) {
            int b = gw;
            float m = -3.4e38f;
            for (int s = lane; s < T_ENC_; s += 32) m = fmaxf(m, ebuf[b * T_ENC_ + s]);
#pragma unroll
            for (int off = 16; off > 0; off >>= 1) m = fmaxf(m, __shfl_xor(m, off, 32));
            float sum = 0.0f;
            for (int s = lane; s < T_ENC_; s += 32) {
                float p = expf(ebuf[b * T_ENC_ + s] - m);
                aw[b * T_ENC_ + s] = p;
                sum += p;
            }
#pragma unroll
            for (int off = 16; off > 0; off >>= 1) sum += __shfl_xor(sum, off, 32);
            float inv = 1.0f / sum;
            for (int s = lane; s < T_ENC_; s += 32) aw[b * T_ENC_ + s] *= inv;
        }
        gridbar(cnt, gen);

        // P3d: ctx = aw @ memory (packed + fp32) ; awc += aw ; emit alignments
        for (int idx = tid; idx < B_ * ENC_DIM_; idx += NTHR_) {
            int b = idx >> 9, d = idx & 511;
            const float* awb = aw + b * T_ENC_;
            const float* mb  = memory + ((size_t)b * T_ENC_) * ENC_DIM_ + d;
            float acc = 0.0f;
            for (int s = 0; s < T_ENC_; ++s) acc += awb[s] * mb[(size_t)s * ENC_DIM_];
            __bf16 cb = (__bf16)acc;
            Aatt[a_index(b, PRE_ + d, NKT_ATT_)] = cb;
            Adec[a_index(b, RNN_ + d, NKT_DEC_)] = cb;
            ctxf[idx] = acc;
        }
        for (int idx = tid; idx < B_ * T_ENC_; idx += NTHR_) {
            float a = aw[idx];
            awc[idx] += a;
            int b = idx / T_ENC_, s = idx - b * T_ENC_;
            out[ALIGN_OFF_ + ((size_t)b * T_DEC_ + t) * T_ENC_ + s] = a;
        }
        gridbar(cnt, gen);

        // P4: decoder LSTM gates = [ah|ctx|dh] @ [Wih|Whh]^T + biases
        wmma_gemm_p<true>(Adec, NKT_DEC_, NKT_DEC_, Wdec, NG_ / 16, gates, NG_,
                          dec_bih, dec_bhh, gw, lane);
        gridbar(cnt, gen);

        // P5: decoder LSTM pointwise -> dh (packed + fp32), dc
        for (int idx = tid; idx < B_ * RNN_; idx += NTHR_) {
            int b = idx >> 10, n = idx & 1023;
            const float* g = gates + (size_t)b * NG_;
            float i = sigf(g[n]), f = sigf(g[RNN_ + n]);
            float gg = tanhf(g[2 * RNN_ + n]), o = sigf(g[3 * RNN_ + n]);
            float c = f * dc[idx] + i * gg;
            dc[idx] = c;
            float h = o * tanhf(c);
            Adec[a_index(b, RNN_ + ENC_DIM_ + n, NKT_DEC_)] = (__bf16)h;
            dhf[idx] = h;
        }
        gridbar(cnt, gen);

        // P6: mel projection + stop gate (fp32 dh/ctx)
        for (int idx = tid; idx < B_ * N_MEL_; idx += NTHR_) {
            int b = idx / N_MEL_, m = idx - b * N_MEL_;
            const float* dh_ = dhf + b * RNN_;
            const float* cx  = ctxf + b * ENC_DIM_;
            const float* w   = projW + (size_t)m * (RNN_ + ENC_DIM_);
            float acc = projB[m];
            for (int j = 0; j < RNN_; ++j) acc += dh_[j] * w[j];
            for (int j = 0; j < ENC_DIM_; ++j) acc += cx[j] * w[RNN_ + j];
            out[((size_t)b * N_MEL_ + m) * T_DEC_ + t] = acc;
        }
        if (tid < B_) {
            int b = tid;
            const float* dh_ = dhf + b * RNN_;
            const float* cx  = ctxf + b * ENC_DIM_;
            float acc = gateB[0];
            for (int j = 0; j < RNN_; ++j) acc += dh_[j] * gateW[j];
            for (int j = 0; j < ENC_DIM_; ++j) acc += cx[j] * gateW[RNN_ + j];
            out[GATE_OFF_ + (size_t)b * T_DEC_ + t] = acc;
        }
        // no barrier needed before next P0 (disjoint buffers; next barrier covers it)
    }
}

// ---------------- host side ----------------
extern "C" void kernel_launch(void* const* d_in, const int* in_sizes, int n_in,
                              void* d_out, int out_size, void* d_ws, size_t ws_size,
                              hipStream_t stream) {
    const float* memory   = (const float*)d_in[0];
    const float* dec_in   = (const float*)d_in[1];
    const int*   lens     = (const int*)d_in[2];
    const float* preW1    = (const float*)d_in[3];
    const float* preW2    = (const float*)d_in[4];
    const float* att_Wih  = (const float*)d_in[5];
    const float* att_Whh  = (const float*)d_in[6];
    const float* att_bih  = (const float*)d_in[7];
    const float* att_bhh  = (const float*)d_in[8];
    const float* q_W      = (const float*)d_in[9];
    const float* m_W      = (const float*)d_in[10];
    const float* v_w      = (const float*)d_in[11];
    const float* loc_conv = (const float*)d_in[12];
    const float* loc_dense= (const float*)d_in[13];
    const float* dec_Wih  = (const float*)d_in[14];
    const float* dec_Whh  = (const float*)d_in[15];
    const float* dec_bih  = (const float*)d_in[16];
    const float* dec_bhh  = (const float*)d_in[17];
    const float* proj_W   = (const float*)d_in[18];
    const float* proj_b   = (const float*)d_in[19];
    const float* gate_W   = (const float*)d_in[20];
    const float* gate_b   = (const float*)d_in[21];
    float* out = (float*)d_out;

    char* ws = (char*)d_ws;
    unsigned* bar = (unsigned*)ws;
    __bf16* stb  = (__bf16*)(ws + 256);                 // state region (bf16 first, 32B aligned)
    float*  xpre = (float*)(ws + 256 + (size_t)STATE_WORDS * 4);
    float*  pm   = xpre + XPRE_N;
    __bf16* Watt = (__bf16*)(pm + PM_N);
    __bf16* Wdec = Watt + WATT_N;
    __bf16* Wq   = Wdec + WDEC_N;

    init_kernel<<<256, 256, 0, stream>>>(bar, (unsigned*)stb, STATE_WORDS);
    prenet_kernel<<<(T_DEC_ * B_) / 8, 256, 0, stream>>>(dec_in, preW1, preW2, xpre);
    pm_kernel<<<PM_N / 256, 256, 0, stream>>>(memory, m_W, pm);
    pack_weights<<<2048, 256, 0, stream>>>(att_Wih, att_Whh, PRE_ + ENC_DIM_, K_ATT_, NG_, Watt);
    pack_weights<<<2048, 256, 0, stream>>>(dec_Wih, dec_Whh, RNN_ + ENC_DIM_, K_DEC_, NG_, Wdec);
    pack_weights<<<64, 256, 0, stream>>>(q_W, q_W, RNN_, RNN_, ATT_DIM_, Wq);
    decoder_kernel<<<NWG_, 256, 0, stream>>>(
        memory, lens, att_bih, att_bhh, v_w, loc_conv, loc_dense,
        dec_bih, dec_bhh, proj_W, proj_b, gate_W, gate_b,
        xpre, pm, Watt, Wdec, Wq, stb, bar, out);
}
